// EncoderBlock_83906481094891
// MI455X (gfx1250) — matrix operand order repaired and checked by reading, not compiled, not run on hardware
//
#include <hip/hip_runtime.h>
#include <hip/hip_bf16.h>
#include <math.h>

// ---------------- problem constants ----------------
constexpr int Bn  = 32;
constexpr int Ln  = 400;
constexpr int Hd  = 512;
constexpr int NHn = 8;
constexpr int Kc  = 7;
constexpr int NCn = 4;
constexpr int DH  = 64;          // Hd / NHn
constexpr int Mrows = Bn * Ln;   // 12800

typedef __attribute__((ext_vector_type(16))) _Float16 v16h;
typedef __attribute__((ext_vector_type(8)))  float    v8f;

// ---------------- WMMA fragment gathers (per CDNA5 ISA layouts) ----
// A matrix 16x32 f16: lanes 0-15 -> M=lane, K = {0..7,16..23}; lanes 16-31 ->
// M=lane-16, K = {8..15,24..31}. base row-major, K contiguous -> 2x b128 each.
__device__ static inline v16h load_frag_a(const _Float16* base, int ld, int lane) {
  int row   = lane & 15;
  int kbase = (lane < 16) ? 0 : 8;
  const _Float16* p = base + row * ld;
  v16h f;
#pragma unroll
  for (int v = 0; v < 8; ++v) {
    int k = kbase + (v & 3) * 2 + (v >> 2) * 16;
    f[2 * v]     = p[k];
    f[2 * v + 1] = p[k + 1];
  }
  return f;
}

// B matrix 32x16 f16, storage N-major / K-contiguous (Bt[n][k]).
// lanes 0-15 -> N=lane, K={0..7,16..23} ; lanes 16-31 -> N=lane-16, K={8..15,24..31}.
__device__ static inline v16h load_frag_b(const _Float16* base, int ld, int lane) {
  int col   = lane & 15;
  int kbase = (lane < 16) ? 0 : 8;
  const _Float16* p = base + col * ld + kbase;
  v16h f;
#pragma unroll
  for (int v = 0; v < 8; ++v) {
    f[2 * v]     = p[2 * v + (v >> 2) * 8];
    f[2 * v + 1] = p[2 * v + 1 + (v >> 2) * 8];
  }
  return f;
}

// ---------------- f32 -> f16 weight conversion ----------------
__global__ __launch_bounds__(256) void cvt_h_kernel(const float* __restrict__ in,
                                                    _Float16* __restrict__ outp) {
  int i = blockIdx.x * 256 + threadIdx.x;
  outp[i] = (_Float16)in[i];
}

// ---------------- positional encoding add ----------------
__global__ __launch_bounds__(256) void add_pe_kernel(const float* __restrict__ x,
                                                     float* __restrict__ out) {
  int idx = blockIdx.x * 256 + threadIdx.x;      // over B*L*H
  int c   = idx & (Hd - 1);
  int row = idx >> 9;
  int l   = row % Ln;
  float f  = __expf(-(float)(c & ~1) * (9.2103403719761836f / (float)Hd));
  float ph = (float)l * f;
  float pe = (c & 1) ? __cosf(ph) : __sinf(ph);
  out[idx] = x[idx] + pe;
}

// ---------------- layernorm over H=512 (f32 in, f16 out) ----------------
__global__ __launch_bounds__(256) void ln_kernel(const float* __restrict__ x,
                                                 const float* __restrict__ g,
                                                 const float* __restrict__ b,
                                                 _Float16* __restrict__ y) {
  __shared__ float red[256];
  size_t row = blockIdx.x;
  const float* xr = x + row * Hd;
  int t = threadIdx.x;
  float v0 = xr[t], v1 = xr[t + 256];
  red[t] = v0 + v1;
  __syncthreads();
  for (int off = 128; off > 0; off >>= 1) {
    if (t < off) red[t] += red[t + off];
    __syncthreads();
  }
  float mean = red[0] * (1.0f / Hd);
  __syncthreads();
  float d0 = v0 - mean, d1 = v1 - mean;
  red[t] = d0 * d0 + d1 * d1;
  __syncthreads();
  for (int off = 128; off > 0; off >>= 1) {
    if (t < off) red[t] += red[t + off];
    __syncthreads();
  }
  float inv = rsqrtf(red[0] * (1.0f / Hd) + 1e-5f);
  y[row * Hd + t]       = (_Float16)(d0 * inv * g[t]       + b[t]);
  y[row * Hd + t + 256] = (_Float16)(d1 * inv * g[t + 256] + b[t + 256]);
}

// ---------------- depthwise conv K=7, same padding (f16 in/out) ------------
__global__ __launch_bounds__(256) void dwconv_kernel(const _Float16* __restrict__ x,
                                                     const float* __restrict__ w,    // (H,1,K)
                                                     const float* __restrict__ bias, // (H)
                                                     _Float16* __restrict__ y) {
  int idx = blockIdx.x * 256 + threadIdx.x;   // over B*L*H
  int c   = idx & (Hd - 1);
  int row = idx >> 9;
  int l   = row % Ln;
  int b   = row / Ln;
  float acc = bias[c];
#pragma unroll
  for (int k = 0; k < Kc; ++k) {
    int ll = l + k - (Kc / 2);
    if (ll >= 0 && ll < Ln)
      acc += (float)x[((size_t)(b * Ln + ll) << 9) + c] * w[c * Kc + k];
  }
  y[idx] = (_Float16)acc;
}

// ---------------- WMMA GEMM: Y = act(X * W^T + bias) (+ residual) ----------
// X: MxK f16, W: NxK f16 (row-major). Block tile 256x64, 8 waves, each wave
// 2 A-frags x 4 B-frags = 8 WMMAs per K-step. K is always Hd (=512).
// Outputs f32 and/or f16.
#define TMg 256
#define TNg 64
#define TKg 32
__global__ __launch_bounds__(256) void gemm_wmma_kernel(
    const _Float16* __restrict__ X, const _Float16* __restrict__ W,
    const float* __restrict__ bias, const float* residual,
    float* Yf, _Float16* Yh, int M, int N, int relu) {
  constexpr int K = Hd;                             // 512, compile-time
  __shared__ __align__(16) _Float16 As[TMg][TKg];   // 16 KB
  __shared__ __align__(16) _Float16 Bs[TNg][TKg];   // 4 KB
  int wave = threadIdx.x >> 5;
  int lane = threadIdx.x & 31;
  int m0 = blockIdx.y * TMg;
  int n0 = blockIdx.x * TNg;

  // per-thread staging slots (loop-invariant): row r0 (+64*ii for A), fixed koff
  int r0 = threadIdx.x >> 2;
  int ko = (threadIdx.x & 3) * 8;
  const _Float16* aPtr = X + (size_t)(m0 + r0) * K + ko;
  const _Float16* bPtr = W + (size_t)(n0 + r0) * K + ko;
  _Float16* aDst = &As[r0][ko];
  _Float16* bDst = &Bs[r0][ko];

  v8f c[2][4] = {};
#pragma unroll 1
  for (int k0 = 0; k0 < K; k0 += TKg) {
    // straight-line staging: 5x global_load_b128 in flight, then LDS stores
    uint4 a0v = *(const uint4*)(aPtr + (size_t)0 * 64 * K);
    uint4 a1v = *(const uint4*)(aPtr + (size_t)1 * 64 * K);
    uint4 a2v = *(const uint4*)(aPtr + (size_t)2 * 64 * K);
    uint4 a3v = *(const uint4*)(aPtr + (size_t)3 * 64 * K);
    uint4 b0v = *(const uint4*)bPtr;
    __builtin_prefetch(aPtr + TKg, 0, 1);           // next K tile of streamed A
    *(uint4*)(aDst + 0 * 64 * TKg) = a0v;
    *(uint4*)(aDst + 1 * 64 * TKg) = a1v;
    *(uint4*)(aDst + 2 * 64 * TKg) = a2v;
    *(uint4*)(aDst + 3 * 64 * TKg) = a3v;
    *(uint4*)bDst = b0v;
    aPtr += TKg;
    bPtr += TKg;
    __syncthreads();

    v16h a0 = load_frag_a(&As[wave * 32][0],      TKg, lane);
    v16h a1 = load_frag_a(&As[wave * 32 + 16][0], TKg, lane);
#pragma unroll
    for (int t = 0; t < 4; ++t) {
      v16h bfrag = load_frag_b(&Bs[t * 16][0], TKg, lane);
      c[0][t] = __builtin_amdgcn_wmma_f32_16x16x32_f16(
          false, a0, false, bfrag, (short)0, c[0][t], false, false);
      c[1][t] = __builtin_amdgcn_wmma_f32_16x16x32_f16(
          false, a1, false, bfrag, (short)0, c[1][t], false, false);
    }
    __syncthreads();
  }

  int col_in = lane & 15;
#pragma unroll
  for (int s = 0; s < 2; ++s) {
    int rbase = m0 + wave * 32 + s * 16 + ((lane < 16) ? 0 : 8);
#pragma unroll
    for (int t = 0; t < 4; ++t) {
      int col  = n0 + t * 16 + col_in;
      float bv = bias[col];
#pragma unroll
      for (int v = 0; v < 8; ++v) {
        size_t idx = (size_t)(rbase + v) * N + col;
        float val = c[s][t][v] + bv;
        if (relu) val = fmaxf(val, 0.0f);
        if (residual) val += residual[idx];
        if (Yf) Yf[idx] = val;
        if (Yh) Yh[idx] = (_Float16)val;
      }
    }
  }
}

// ---------------- flash attention: one wave per (b,h,q-tile of 16) ---------
// qkv is f16 (B*L, 3H); ctx written f16 (B*L, H).
__global__ __launch_bounds__(32) void attn_kernel(const _Float16* __restrict__ qkv,
                                                  const int* __restrict__ mask,
                                                  _Float16* __restrict__ ctx) {
  __shared__ __align__(16) _Float16 Qs[16][64];
  __shared__ __align__(16) _Float16 Ks[16][64];  // B layout for Q*K^T
  __shared__ __align__(16) _Float16 Vs[64][32];  // B layout for P*V (keys 16..31 zero)
  __shared__ __align__(16) _Float16 Ps[16][32];  // A layout for P*V (cols 16..31 zero)
  __shared__ float Ss[16][16];
  __shared__ float alphaS[16];
  __shared__ float lS[16];

  int bh = blockIdx.x;            // b*NH + h
  int qt = blockIdx.y;            // 0..L/16-1
  int b  = bh >> 3;
  int h  = bh & 7;
  int lane = threadIdx.x;
  int colS = lane & 15;
  int rb   = (lane < 16) ? 0 : 8;

  // zero the pad regions once
  for (int i = lane; i < 64 * 16; i += 32) Vs[i >> 4][16 + (i & 15)] = (_Float16)0.0f;
  for (int i = lane; i < 16 * 16; i += 32) Ps[i >> 4][16 + (i & 15)] = (_Float16)0.0f;

  size_t qrow0 = (size_t)(b * Ln + qt * 16);
  // Q tile copy: 128 x uint4 (b128), 4 per lane
  for (int i = lane; i < 128; i += 32) {
    int r = i >> 3, d8 = (i & 7) * 8;
    *(uint4*)&Qs[r][d8] =
        *(const uint4*)&qkv[(qrow0 + r) * (3 * Hd) + h * DH + d8];
  }
  __syncthreads();
  v16h qa0 = load_frag_a(&Qs[0][0], 64, lane);
  v16h qa1 = load_frag_a(&Qs[0][32], 64, lane);

  float mrow = -INFINITY, lrow = 0.0f;  // lanes 0..15 own row = lane
  v8f o[4] = {};
  const float scale = 0.125f;           // 1/sqrt(64), applied to scores

  for (int kt = 0; kt < Ln / 16; ++kt) {
    size_t krow0 = (size_t)(b * Ln + kt * 16);
    for (int i = lane; i < 128; i += 32) {
      int r = i >> 3, d8 = (i & 7) * 8;
      const _Float16* base = &qkv[(krow0 + r) * (3 * Hd) + h * DH + d8];
      *(uint4*)&Ks[r][d8] = *(const uint4*)(base + Hd);   // K tile (b128 copy)
      const _Float16* vsrc = base + 2 * Hd;               // V: transpose scatter
#pragma unroll
      for (int j = 0; j < 8; ++j) Vs[d8 + j][r] = vsrc[j];
    }
    __syncthreads();

    v8f s = {};
    s = __builtin_amdgcn_wmma_f32_16x16x32_f16(false, qa0, false,
        load_frag_b(&Ks[0][0], 64, lane), (short)0, s, false, false);
    s = __builtin_amdgcn_wmma_f32_16x16x32_f16(false, qa1, false,
        load_frag_b(&Ks[0][32], 64, lane), (short)0, s, false, false);
#pragma unroll
    for (int v = 0; v < 8; ++v) Ss[rb + v][colS] = s[v] * scale;
    __syncthreads();

    if (lane < 16) {
      float sv[16];
      float mx = mrow;
#pragma unroll
      for (int j = 0; j < 16; ++j) {
        float val = Ss[lane][j];
        if (mask[b * Ln + kt * 16 + j] == 0) val = -1e9f;
        sv[j] = val;
        mx = fmaxf(mx, val);
      }
      float alpha = __expf(mrow - mx);
      float sum = 0.0f;
#pragma unroll
      for (int j = 0; j < 16; ++j) {
        float p = __expf(sv[j] - mx);
        sum += p;
        Ps[lane][j] = (_Float16)p;
      }
      lrow = lrow * alpha + sum;
      mrow = mx;
      alphaS[lane] = alpha;
    }
    __syncthreads();

#pragma unroll
    for (int v = 0; v < 8; ++v) {
      float al = alphaS[rb + v];
#pragma unroll
      for (int t = 0; t < 4; ++t) o[t][v] *= al;
    }
    v16h pa = load_frag_a(&Ps[0][0], 32, lane);
#pragma unroll
    for (int t = 0; t < 4; ++t) {
      v16h vb = load_frag_b(&Vs[t * 16][0], 32, lane);
      o[t] = __builtin_amdgcn_wmma_f32_16x16x32_f16(
          false, pa, false, vb, (short)0, o[t], false, false);
    }
    __syncthreads();
  }

  if (lane < 16) lS[lane] = 1.0f / lrow;
  __syncthreads();
#pragma unroll
  for (int t = 0; t < 4; ++t)
#pragma unroll
    for (int v = 0; v < 8; ++v) {
      int row = rb + v;
      int d   = t * 16 + colS;
      ctx[(qrow0 + row) * Hd + h * DH + d] = (_Float16)(o[t][v] * lS[row]);
    }
}

// ---------------- host launch ----------------
extern "C" void kernel_launch(void* const* d_in, const int* in_sizes, int n_in,
                              void* d_out, int out_size, void* d_ws, size_t ws_size,
                              hipStream_t stream) {
  (void)in_sizes; (void)n_in; (void)out_size; (void)ws_size;

  const float* x         = (const float*)d_in[0];
  const int*   mask      = (const int*)d_in[1];
  const float* conv_ln_g = (const float*)d_in[2];
  const float* conv_ln_b = (const float*)d_in[3];
  const float* dw_w      = (const float*)d_in[4];
  const float* dw_b      = (const float*)d_in[5];
  const float* pw_w      = (const float*)d_in[6];
  const float* pw_b      = (const float*)d_in[7];
  const float* att_ln_g  = (const float*)d_in[8];
  const float* att_ln_b  = (const float*)d_in[9];
  const float* qkv_w     = (const float*)d_in[10];
  const float* qkv_b     = (const float*)d_in[11];
  const float* out_w     = (const float*)d_in[12];
  const float* out_b     = (const float*)d_in[13];
  const float* ff_ln_g   = (const float*)d_in[14];
  const float* ff_ln_b   = (const float*)d_in[15];
  const float* ff_w      = (const float*)d_in[16];
  const float* ff_b      = (const float*)d_in[17];

  float*    out = (float*)d_out;               // running activation (B*L, H) f32
  _Float16* hws = (_Float16*)d_ws;
  size_t MH = (size_t)Mrows * Hd;

  _Float16* norm_h = hws;                      // (B*L, H)
  _Float16* dwb_h  = norm_h + MH;              // (B*L, H): dwconv out / attn ctx
  _Float16* qkv_h  = dwb_h + MH;               // (B*L, 3H)
  _Float16* wpw_h  = qkv_h + 3 * MH;           // NC*H*H
  _Float16* wqkv_h = wpw_h + (size_t)NCn * Hd * Hd;
  _Float16* wout_h = wqkv_h + (size_t)3 * Hd * Hd;
  _Float16* wff_h  = wout_h + (size_t)Hd * Hd;

  // one-time (per launch) f16 weight conversion
  cvt_h_kernel<<<NCn * Hd * Hd / 256, 256, 0, stream>>>(pw_w, wpw_h);
  cvt_h_kernel<<<3 * Hd * Hd / 256, 256, 0, stream>>>(qkv_w, wqkv_h);
  cvt_h_kernel<<<Hd * Hd / 256, 256, 0, stream>>>(out_w, wout_h);
  cvt_h_kernel<<<Hd * Hd / 256, 256, 0, stream>>>(ff_w, wff_h);

  dim3 gemmHH(Hd / TNg, Mrows / TMg);          // (8, 50)
  dim3 gemmQKV(3 * Hd / TNg, Mrows / TMg);     // (24, 50)

  add_pe_kernel<<<(int)(MH / 256), 256, 0, stream>>>(x, out);

  for (int i = 0; i < NCn; ++i) {
    ln_kernel<<<Mrows, 256, 0, stream>>>(out, conv_ln_g + i * Hd, conv_ln_b + i * Hd, norm_h);
    dwconv_kernel<<<(int)(MH / 256), 256, 0, stream>>>(norm_h, dw_w + i * Hd * Kc,
                                                       dw_b + i * Hd, dwb_h);
    gemm_wmma_kernel<<<gemmHH, 256, 0, stream>>>(dwb_h, wpw_h + (size_t)i * Hd * Hd,
                                                 pw_b + i * Hd, out, out, nullptr,
                                                 Mrows, Hd, 1);
  }

  // attention
  ln_kernel<<<Mrows, 256, 0, stream>>>(out, att_ln_g, att_ln_b, norm_h);
  gemm_wmma_kernel<<<gemmQKV, 256, 0, stream>>>(norm_h, wqkv_h, qkv_b, nullptr,
                                                nullptr, qkv_h, Mrows, 3 * Hd, 0);
  attn_kernel<<<dim3(Bn * NHn, Ln / 16), 32, 0, stream>>>(qkv_h, mask, dwb_h);
  gemm_wmma_kernel<<<gemmHH, 256, 0, stream>>>(dwb_h, wout_h, out_b, out, out,
                                               nullptr, Mrows, Hd, 0);

  // feed-forward
  ln_kernel<<<Mrows, 256, 0, stream>>>(out, ff_ln_g, ff_ln_b, norm_h);
  gemm_wmma_kernel<<<gemmHH, 256, 0, stream>>>(norm_h, wff_h, ff_b, out, out,
                                               nullptr, Mrows, Hd, 1);
}
